// quadratic_kappa_55808805044854
// MI455X (gfx1250) — 8-bit matrix operand order repaired and checked by reading, compiled, not run on hardware
//
#include <hip/hip_runtime.h>


typedef __attribute__((ext_vector_type(8))) int v8i;

__global__ __launch_bounds__(256) void qwk_wmma_kernel(
    const int* __restrict__ y_pred,
    const int* __restrict__ y_gt,
    float* __restrict__ out,
    int N) {
  __shared__ int conf[256];

  const int tid = threadIdx.x;
  conf[tid] = 0;
  __syncthreads();

  const int lane = tid & 31;
  const int wave = tid >> 5;
  const int row  = lane & 15;
  const int hi   = lane >> 4;

  v8i acc = {};

  const int samplesPerWave = N >> 3;
  const int base = wave * samplesPerWave;

  for (int chunk = 0; chunk < samplesPerWave; chunk += 64) {
    const int* gt = y_gt  + base + chunk;
    const int* pr = y_pred + base + chunk;
    v8i a, b;
#pragma unroll
    for (int v = 0; v < 8; ++v) {
      {
        const int ka = ((v >> 1) << 4) + ((v & 1) << 2) + (hi << 3);
        const int4 g = *reinterpret_cast<const int4*>(gt + ka);
        unsigned av = (unsigned)(g.x == row)
                    | ((unsigned)(g.y == row) << 8)
                    | ((unsigned)(g.z == row) << 16)
                    | ((unsigned)(g.w == row) << 24);
        a[v] = (int)av;
      }
      {
        const int kb = ((v >> 1) << 4) + ((v & 1) << 2) + (hi << 3);
        const int4 p = *reinterpret_cast<const int4*>(pr + kb);
        unsigned bv = (unsigned)(p.x == row)
                    | ((unsigned)(p.y == row) << 8)
                    | ((unsigned)(p.z == row) << 16)
                    | ((unsigned)(p.w == row) << 24);
        b[v] = (int)bv;
      }
    }
    acc = __builtin_amdgcn_wmma_i32_16x16x64_iu8(
         false, a,  false, b, acc,
         false,  false);
  }

#pragma unroll
  for (int r = 0; r < 8; ++r) {
    const int m = r + (hi << 3);
    atomicAdd(&conf[m * 16 + row], acc[r]);
  }
  __syncthreads();

  if (tid == 0) {
    double gh[16], ph[16];
    for (int i = 0; i < 16; ++i) { gh[i] = 0.0; ph[i] = 0.0; }
    double num_s = 0.0;
    for (int i = 0; i < 16; ++i) {
      for (int j = 0; j < 16; ++j) {
        const double c = (double)conf[i * 16 + j];
        const double d = (double)(i - j);
        num_s += d * d * c;
        gh[i] += c;
        ph[j] += c;
      }
    }
    double den_s = 0.0;
    for (int i = 0; i < 16; ++i) {
      for (int j = 0; j < 16; ++j) {
        const double d = (double)(i - j);
        den_s += d * d * gh[i] * ph[j];
      }
    }
    out[0] = (float)(1.0 - (double)N * num_s / den_s);
  }
}

extern "C" void kernel_launch(void* const* d_in, const int* in_sizes, int n_in,
                              void* d_out, int out_size, void* d_ws, size_t ws_size,
                              hipStream_t stream) {
  const int* y_pred = (const int*)d_in[0];
  const int* y_gt   = (const int*)d_in[1];
  float* out        = (float*)d_out;
  const int N       = in_sizes[0];
  qwk_wmma_kernel<<<1, 256, 0, stream>>>(y_pred, y_gt, out, N);
}
